// quantize_linearlayer_multimode_55104430407989
// MI455X (gfx1250) — matrix operand order repaired and checked by reading, not compiled, not run on hardware
//
#include <hip/hip_runtime.h>
#include <hip/hip_bf16.h>

typedef __attribute__((ext_vector_type(16))) __bf16 v16bf;
typedef __attribute__((ext_vector_type(8)))  float  v8f;

struct U32B { uint4 a, b; };   // 32 bytes == v16bf

__device__ __forceinline__ unsigned short f32_to_bf16(float f) {
    unsigned u = __builtin_bit_cast(unsigned, f);
    unsigned r = u + 0x7fffu + ((u >> 16) & 1u);   // round-to-nearest-even
    return (unsigned short)(r >> 16);
}

// Pack two f32 -> one dword of 2x bf16 using only 32-bit VALU ops
// (avoids clang lowering through v_mov_b16 sub-register moves).
__device__ __forceinline__ unsigned pack_bf16x2(float lo, float hi) {
    unsigned ul = __builtin_bit_cast(unsigned, lo);
    unsigned uh = __builtin_bit_cast(unsigned, hi);
    unsigned rl = ul + 0x7fffu + ((ul >> 16) & 1u);
    unsigned rh = uh + 0x7fffu + ((uh >> 16) & 1u);
    return (rl >> 16) | (rh & 0xffff0000u);
}

// ---------------------------------------------------------------------------
// Kernel 1: groupwise grid quantization  Wq = (1-x)*down + x*up  -> bf16
// One wave32 per group; grid-stride over groups. groupsize/wbits read on-device.
// ---------------------------------------------------------------------------
__global__ void quant_wq_kernel(const float* __restrict__ W,
                                const float* __restrict__ x,
                                const int* __restrict__ wbits_p,
                                const int* __restrict__ gs_p,
                                unsigned short* __restrict__ wq,
                                int out_f, int in_f) {
    const int gs = gs_p[0];
    const int wb = wbits_p[0];
    const float levels = (float)((1 << wb) - 1);

    const int lane = threadIdx.x & 31;
    const int wave = threadIdx.x >> 5;
    const int wavesPerBlock = blockDim.x >> 5;

    const long G = in_f / gs;
    const long totalGroups = (long)out_f * G;
    long g = (long)blockIdx.x * wavesPerBlock + wave;
    const long stride = (long)gridDim.x * wavesPerBlock;

    for (; g < totalGroups; g += stride) {
        const long row = g / G;
        const long grp = g % G;
        const float* wrow = W  + row * in_f + grp * gs;
        const float* xrow = x  + row * in_f + grp * gs;
        unsigned short* orow = wq + row * in_f + grp * gs;

        float mn =  3.4e38f, mx = -3.4e38f;
        for (int e = lane; e < gs; e += 32) {
            float v = wrow[e];
            mn = fminf(mn, v);
            mx = fmaxf(mx, v);
        }
        #pragma unroll
        for (int off = 16; off > 0; off >>= 1) {
            mn = fminf(mn, __shfl_xor(mn, off, 32));
            mx = fmaxf(mx, __shfl_xor(mx, off, 32));
        }
        const float scale = (mx - mn) / levels;
        const float safe  = (scale == 0.0f) ? 1.0f : scale;

        for (int e = lane; e < gs; e += 32) {
            float v  = wrow[e];
            float q  = (v - mn) / safe;
            float kd = floorf(q);
            float ku = fminf(kd + 1.0f, levels);
            float dn = kd * scale + mn;
            float up = ku * scale + mn;
            float xx = xrow[e];
            float w  = (1.0f - xx) * dn + xx * up;
            orow[e]  = f32_to_bf16(w);
        }
    }
}

// ---------------------------------------------------------------------------
// Kernel 2: out[M,N] = A[M,K](f32->bf16) @ Wq[N,K]^T + bias, WMMA bf16 path.
// Block: 256 threads (8 waves), 128x128 tile, BK=64, double-buffered LDS.
// A tile: global f32 -> VGPR convert -> LDS.  B tile: global_load_async_to_lds.
// Wave grid 4(m) x 2(n); each wave: 32x64 = 2x4 WMMA 16x16 accumulators.
// ---------------------------------------------------------------------------
#define BM 128
#define BN 128
#define BK 64
#define LDA 72   // padded LDS row stride in halves (144B: 16B aligned, skews banks)

__global__ __launch_bounds__(256) void gemm_bf16_wmma_kernel(
        const float* __restrict__ A,
        const unsigned short* __restrict__ Bw,
        const float* __restrict__ bias,
        float* __restrict__ out,
        int M, int N, int K) {
    __shared__ unsigned short sA[2][BM * LDA];
    __shared__ unsigned short sB[2][BN * LDA];

    const int tid  = threadIdx.x;
    const int lane = tid & 31;
    const int wave = tid >> 5;
    const int wm   = wave & 3;        // 0..3 -> m offset wm*32
    const int wn   = wave >> 2;       // 0..1 -> n offset wn*64

    const int blockM = blockIdx.y * BM;
    const int blockN = blockIdx.x * BN;

    v8f acc[2][4] = {};

    // cooperative-load coordinates: 2 threads per row, 32 halves each
    const int ldrow = tid >> 1;            // 0..127
    const int ldseg = (tid & 1) * 32;      // half-element offset in K tile

    const float*          aptr   = A  + (long)(blockM + ldrow) * K + ldseg;
    const unsigned short* bgbase = Bw + (long)(blockN + ldrow) * K + ldseg;

    // fragment coordinates
    const int r15   = lane & 15;
    const int hselA = (lane >> 4) * 16;   // A: half-wave contiguous K
    const int hselB = (lane >> 4) * 16;   // B: half-wave contiguous K

    // ---- helpers as macros (keep everything in-scope) ----
    #define ASYNC_B_TILE(KB, BUF)                                                   \
        do {                                                                        \
            _Pragma("unroll")                                                       \
            for (int i_ = 0; i_ < 4; ++i_) {                                        \
                unsigned lds_off_ = (unsigned)(uintptr_t)                           \
                    &sB[BUF][ldrow * LDA + ldseg + i_ * 8];                         \
                unsigned long long ga_ = (unsigned long long)(uintptr_t)            \
                    (bgbase + (KB) + i_ * 8);                                       \
                asm volatile("global_load_async_to_lds_b128 %0, %1, off"            \
                             :: "v"(lds_off_), "v"(ga_) : "memory");                \
            }                                                                       \
        } while (0)

    #define STORE_A_TILE(AF, BUF)                                                  \
        do {                                                                        \
            unsigned pa_[16];                                                       \
            _Pragma("unroll")                                                       \
            for (int i_ = 0; i_ < 8; ++i_) {                                        \
                pa_[2*i_+0] = pack_bf16x2(AF[i_].x, AF[i_].y);                      \
                pa_[2*i_+1] = pack_bf16x2(AF[i_].z, AF[i_].w);                      \
            }                                                                       \
            uint4* sa_ = (uint4*)&sA[BUF][ldrow * LDA + ldseg];                     \
            _Pragma("unroll")                                                       \
            for (int i_ = 0; i_ < 4; ++i_) {                                        \
                uint4 u_; u_.x = pa_[4*i_]; u_.y = pa_[4*i_+1];                     \
                u_.z = pa_[4*i_+2]; u_.w = pa_[4*i_+3];                             \
                sa_[i_] = u_;                                                       \
            }                                                                       \
        } while (0)

    // ---- prologue: stage tile kb=0 into buffer 0 ----
    {
        float4 af0[8];
        #pragma unroll
        for (int i = 0; i < 8; ++i)
            af0[i] = *(const float4*)(aptr + i * 4);
        ASYNC_B_TILE(0, 0);
        STORE_A_TILE(af0, 0);
        asm volatile("s_wait_asynccnt 0x0" ::: "memory");
    }
    __syncthreads();

    int buf = 0;
    for (int kb = 0; kb < K; kb += BK) {
        const bool hasNext = (kb + BK) < K;

        // issue next tile's global traffic first: it overlaps the WMMA block
        float4 af[8];
        if (hasNext) {
            #pragma unroll
            for (int i = 0; i < 8; ++i)
                af[i] = *(const float4*)(aptr + kb + BK + i * 4);
            ASYNC_B_TILE(kb + BK, buf ^ 1);
            if (kb + 2 * BK < K)
                __builtin_prefetch(aptr + kb + 2 * BK, 0, 1);
        }

        // ---- compute on current buffer: 2 K-slices of 32, 8 WMMAs each ----
        const unsigned short* As = sA[buf];
        const unsigned short* Bs = sB[buf];
        #pragma unroll
        for (int kk = 0; kk < BK; kk += 32) {
            v16bf afr[2], bfr[4];
            #pragma unroll
            for (int i = 0; i < 2; ++i) {
                const unsigned short* p = &As[(wm * 32 + i * 16 + r15) * LDA + kk + hselA];
                U32B u; u.a = *(const uint4*)p; u.b = *(const uint4*)(p + 8);
                afr[i] = __builtin_bit_cast(v16bf, u);
            }
            #pragma unroll
            for (int j = 0; j < 4; ++j) {
                const unsigned short* p = &Bs[(wn * 64 + j * 16 + r15) * LDA + kk + hselB];
                U32B u; u.a = *(const uint4*)p; u.b = *(const uint4*)(p + 8);
                bfr[j] = __builtin_bit_cast(v16bf, u);
            }
            #pragma unroll
            for (int i = 0; i < 2; ++i)
                #pragma unroll
                for (int j = 0; j < 4; ++j)
                    acc[i][j] = __builtin_amdgcn_wmma_f32_16x16x32_bf16(
                        false, afr[i], false, bfr[j],
                        (short)0, acc[i][j], false, false);
        }

        // ---- finish staging next buffer, then one barrier per iteration ----
        if (hasNext) {
            STORE_A_TILE(af, buf ^ 1);
            asm volatile("s_wait_asynccnt 0x0" ::: "memory");
        }
        __syncthreads();
        buf ^= 1;
    }

    // ---- epilogue: C layout = 8 VGPRs; lanes 0-15 -> M=r, lanes 16-31 -> M=8+r ----
    const int hs = (lane >> 4) * 8;
    #pragma unroll
    for (int j = 0; j < 4; ++j) {
        const int col = blockN + wn * 64 + j * 16 + r15;
        const float bv = bias[col];
        #pragma unroll
        for (int i = 0; i < 2; ++i) {
            const int rowBase = blockM + wm * 32 + i * 16 + hs;
            #pragma unroll
            for (int rr = 0; rr < 8; ++rr) {
                out[(long)(rowBase + rr) * N + col] = acc[i][j][rr] + bv;
            }
        }
    }
    #undef ASYNC_B_TILE
    #undef STORE_A_TILE
}

// ---------------------------------------------------------------------------
extern "C" void kernel_launch(void* const* d_in, const int* in_sizes, int n_in,
                              void* d_out, int out_size, void* d_ws, size_t ws_size,
                              hipStream_t stream) {
    const float* input = (const float*)d_in[0];
    const float* x     = (const float*)d_in[1];
    const float* W     = (const float*)d_in[2];
    const float* bias  = (const float*)d_in[3];
    const int* wbits   = (const int*)d_in[4];
    const int* gsz     = (const int*)d_in[5];

    const int out_f  = in_sizes[3];            // 4096
    const int in_f   = in_sizes[1] / out_f;    // 4096
    const int tokens = in_sizes[0] / in_f;     // 8192

    unsigned short* wq = (unsigned short*)d_ws; // bf16 Wq [out_f, in_f] (32 MB)

    // Phase 1: quantize W -> bf16 Wq (grid-stride, groupsize read on device)
    quant_wq_kernel<<<4096, 256, 0, stream>>>(W, x, wbits, gsz, wq, out_f, in_f);

    // Phase 2: WMMA GEMM  out = input @ Wq^T + bias
    dim3 grid(out_f / BN, tokens / BM);
    gemm_bf16_wmma_kernel<<<grid, 256, 0, stream>>>(input, wq, bias, (float*)d_out,
                                                    tokens, out_f, in_f);
}
